// Net_10514079940902
// MI455X (gfx1250) — matrix operand order repaired and checked by reading, not compiled, not run on hardware
//
#include <hip/hip_runtime.h>

// ---------------------------------------------------------------- constants
#define BB   1024      // batch
#define TT   128       // time steps
#define SS   10        // branches
#define HH   64        // hidden
#define H4   256       // 4*H
#define BT   64        // batch tile per workgroup
#define NTILES (BB/BT) // 16

#define WTLD 66        // K-leading-dim for f16 weight tiles in LDS (132B rows, conflict-free)
#define HLD  72        // K-leading-dim for f16 activation tiles (144B rows: 16B-aligned, conflict-free)
#define GLD  260       // N-leading-dim for f32 gate buffer in LDS

typedef _Float16 h16;
typedef __attribute__((ext_vector_type(2)))  _Float16 h16x2;
typedef __attribute__((ext_vector_type(16))) _Float16 v16h;
typedef __attribute__((ext_vector_type(8)))  float    v8f;

#define WMMA(A, Bf, C) \
  __builtin_amdgcn_wmma_f32_16x16x32_f16(false, (A), false, (Bf), (short)0, (C), false, false)

// ---------------------------------------------------------------- helpers
static __device__ __forceinline__ float sigf(float x) {
  x = fminf(fmaxf(x, -20.f), 20.f);
  return 1.f / (1.f + __expf(-x));
}
static __device__ __forceinline__ float tanhfast(float x) {
  x = fminf(fmaxf(x, -15.f), 15.f);
  float e = __expf(2.f * x);
  return (e - 1.f) / (e + 1.f);
}

static __device__ __forceinline__ void wait_async0() {
#if __has_builtin(__builtin_amdgcn_s_wait_asynccnt)
  __builtin_amdgcn_s_wait_asynccnt(0);
#else
  asm volatile("s_wait_asynccnt 0x0" ::: "memory");
#endif
}

// async copy of one 16B chunk: global -> LDS (ASYNCcnt-tracked)
static __device__ __forceinline__ void async_b128(unsigned lds_addr, const void* gaddr) {
  asm volatile("global_load_async_to_lds_b128 %0, %1, off"
               :: "v"(lds_addr), "v"((unsigned long long)(size_t)gaddr)
               : "memory");
}
static __device__ __forceinline__ void async_b32(unsigned lds_addr, const void* gaddr) {
  asm volatile("global_load_async_to_lds_b32 %0, %1, off"
               :: "v"(lds_addr), "v"((unsigned long long)(size_t)gaddr)
               : "memory");
}
static __device__ __forceinline__ unsigned lds_off(const void* p) {
  return (unsigned)(size_t)p;  // low 32 bits of generic address = LDS offset
}

// issue async copy of a contiguous [64 x 64] f16 global tile (8192B) into a
// padded [64][HLD] f16 LDS tile; 2 x b128 per thread (256 threads).
static __device__ __forceinline__ void issue_tile_async(const h16* __restrict__ gsrc,
                                                        h16* __restrict__ ldst, int tid) {
#pragma unroll
  for (int r = 0; r < 2; ++r) {
    int e16 = r * 256 + tid;        // which 16-byte chunk (0..511)
    int row = e16 >> 3;             // 8 chunks per 128B row
    int col = (e16 & 7) * 16;       // byte offset within row
    async_b128(lds_off(ldst) + row * (HLD * 2) + col, (const char*)gsrc + e16 * 16);
  }
}

// A fragment: 16x32 f16 tile (M rows x K). Lanes 0-15 -> M=0..15; element s holds
// K = s (lanes 0-15) or K = 16 + s (lanes 16-31).
static __device__ __forceinline__ v16h load_afrag(const h16* __restrict__ a, int ld,
                                                  int m0, int k0, int lane) {
  v16h r;
  const int m  = m0 + (lane & 15);
  const int hi = (lane >> 4) & 1;
  const h16* p = a + m * ld + k0;
#pragma unroll
  for (int j = 0; j < 8; ++j) {
    int kk = j * 2 + hi * 16;
    r[2 * j]     = p[kk];
    r[2 * j + 1] = p[kk + 1];
  }
  return r;
}

// B fragment: 32x16 f16 tile (K x N), weights stored transposed in LDS: wt[n][k].
static __device__ __forceinline__ v16h load_bfrag(const h16* __restrict__ wt, int ld,
                                                  int n0, int k0, int lane) {
  v16h r;
  const int n  = n0 + (lane & 15);
  const int hi = (lane >> 4) & 1;
  const h16* p = wt + n * ld + k0 + hi * 16;
#pragma unroll
  for (int v = 0; v < 8; ++v) {
    r[2 * v]     = p[2 * v];
    r[2 * v + 1] = p[2 * v + 1];
  }
  return r;
}

// C/D fragment: 16x16 f32. VGPR r: lanes 0-15 -> M=r, lanes 16-31 -> M=8+r.
static __device__ __forceinline__ void store_cfrag(float* __restrict__ g, int ld,
                                                   int m0, int n0, int lane, v8f c) {
  const int n  = n0 + (lane & 15);
  const int hi = (lane >> 4) & 1;
#pragma unroll
  for (int rr = 0; rr < 8; ++rr) g[(m0 + rr + hi * 8) * ld + n] = c[rr];
}

// generic LDS-resident matmul:  out[rows][ncols] = A1@W(base1) (+ A2@W(base2))
static __device__ __forceinline__ void mm_block(const h16* __restrict__ wAll, int base1,
                                                const h16* __restrict__ A1, int base2,
                                                const h16* __restrict__ A2,
                                                float* __restrict__ gdst, int gld,
                                                int rows, int ncols, int wave, int lane) {
  const int ntiles = ncols >> 4;
  const int mtiles = rows >> 4;
  const int total  = ntiles * mtiles;
  for (int tile = wave; tile < total; tile += 8) {
    const int nt = tile % ntiles;
    const int mt = tile / ntiles;
    v8f c = {0, 0, 0, 0, 0, 0, 0, 0};
    {
      v16h a0 = load_afrag(A1, HLD, mt * 16, 0, lane);
      v16h a1 = load_afrag(A1, HLD, mt * 16, 32, lane);
      v16h b0 = load_bfrag(wAll, WTLD, base1 + nt * 16, 0, lane);
      v16h b1 = load_bfrag(wAll, WTLD, base1 + nt * 16, 32, lane);
      c = WMMA(a0, b0, c);
      c = WMMA(a1, b1, c);
    }
    if (A2) {
      v16h a0 = load_afrag(A2, HLD, mt * 16, 0, lane);
      v16h a1 = load_afrag(A2, HLD, mt * 16, 32, lane);
      v16h b0 = load_bfrag(wAll, WTLD, base2 + nt * 16, 0, lane);
      v16h b1 = load_bfrag(wAll, WTLD, base2 + nt * 16, 32, lane);
      c = WMMA(a0, b0, c);
      c = WMMA(a1, b1, c);
    }
    store_cfrag(gdst, gld, mt * 16, nt * 16, lane, c);
  }
}

// ---------------------------------------------------------------- utility kernels
__global__ void zero_kernel(float* __restrict__ p, int n) {
  for (int i = blockIdx.x * blockDim.x + threadIdx.x; i < n; i += gridDim.x * blockDim.x)
    p[i] = 0.f;
}

__global__ void cvt_f16_kernel(const float* __restrict__ src, h16* __restrict__ dst,
                               int npairs) {
  for (int i = blockIdx.x * blockDim.x + threadIdx.x; i < npairs;
       i += gridDim.x * blockDim.x) {
    h16x2 v;
    v[0] = (h16)src[2 * i];
    v[1] = (h16)src[2 * i + 1];
    *(h16x2*)(dst + 2 * i) = v;
  }
}

// ---------------------------------------------------------------- phase A: 21 scalar-input LSTM scans
// grid.x = batch tile (16), grid.y = stream (0=Y, 1..10 = Xp branch, 11..20 = Xn branch)
#define A_OFF_WT 0
#define A_OFF_G  33792
#define A_OFF_HB 100352
#define A_OFF_XB 109568
#define A_OFF_WB 110080
#define A_OFF_BB 111104
#define SMEM_A   112128
__global__ void lstm_scan_kernel(const float* __restrict__ Y, const float* __restrict__ Xp,
                                 const float* __restrict__ Xn, const float* __restrict__ Wy,
                                 const float* __restrict__ Uy, const float* __restrict__ by,
                                 const float* __restrict__ Wp, const float* __restrict__ Up,
                                 const float* __restrict__ bp, const float* __restrict__ Wn,
                                 const float* __restrict__ Un, const float* __restrict__ bn,
                                 float* __restrict__ yseq, float* __restrict__ Pt,
                                 float* __restrict__ Nt) {
  extern __shared__ char smem[];
  h16*   wt = (h16*)(smem + A_OFF_WT);   // [H4][WTLD]  U transposed, f16
  float* g  = (float*)(smem + A_OFF_G);  // [BT][GLD]   gates f32
  h16*   hb = (h16*)(smem + A_OFF_HB);   // [BT][HLD]   hidden f16
  float* xb = (float*)(smem + A_OFF_XB); // [2][BT]     double-buffered x_t
  float* Wb = (float*)(smem + A_OFF_WB); // [H4] input weight (rank-1)
  float* bb = (float*)(smem + A_OFF_BB); // [H4] bias

  const int tid = threadIdx.x, lane = tid & 31, wave = tid >> 5;
  const int b0  = blockIdx.x * BT;
  const int s   = blockIdx.y;

  const float *Wv, *Um, *bv;
  if (s == 0) {
    Wv = Wy; Um = Uy; bv = by;
  } else if (s <= SS) {
    int si = s - 1;
    Wv = Wp + si * H4; Um = Up + (size_t)si * HH * H4; bv = bp + si * H4;
  } else {
    int si = s - 1 - SS;
    Wv = Wn + si * H4; Um = Un + (size_t)si * HH * H4; bv = bn + si * H4;
  }
  float* outp = (s == 0) ? yseq : (s <= SS ? Pt : Nt);

  for (int idx = tid; idx < HH * H4; idx += 256) {
    int k = idx >> 8, n = idx & 255;
    wt[n * WTLD + k] = (h16)Um[idx];
  }
  for (int idx = tid; idx < H4; idx += 256) { Wb[idx] = Wv[idx]; bb[idx] = bv[idx]; }
  for (int idx = tid; idx < BT * HLD; idx += 256) hb[idx] = (h16)0.f;
  float creg[16];
#pragma unroll
  for (int i = 0; i < 16; ++i) creg[i] = 0.f;

  // prologue: async-gather x for t=0 into xb[0]
  if (tid < BT) {
    int b = b0 + tid;
    const float* gp = (s == 0) ? &Y[(size_t)b * TT]
                    : (s <= SS) ? &Xp[((size_t)b * TT) * SS + (s - 1)]
                                : &Xn[((size_t)b * TT) * SS + (s - 1 - SS)];
    async_b32(lds_off(xb + tid), gp);
  }
  __syncthreads();

  // each wave owns two N-tiles of the [64,256] output; B fragments cached in VGPRs
  v16h bf00 = load_bfrag(wt, WTLD, (wave * 2 + 0) * 16, 0, lane);
  v16h bf01 = load_bfrag(wt, WTLD, (wave * 2 + 0) * 16, 32, lane);
  v16h bf10 = load_bfrag(wt, WTLD, (wave * 2 + 1) * 16, 0, lane);
  v16h bf11 = load_bfrag(wt, WTLD, (wave * 2 + 1) * 16, 32, lane);

#pragma unroll 1
  for (int t = 0; t < TT; ++t) {
    wait_async0();
    __syncthreads();  // x[t] ready; previous consume done (hb stable, g free)
    const int cur = t & 1;
    if (t + 1 < TT && tid < BT) {  // prefetch x for t+1
      int b = b0 + tid;
      int tn = t + 1;
      const float* gp = (s == 0) ? &Y[(size_t)b * TT + tn]
                      : (s <= SS) ? &Xp[((size_t)b * TT + tn) * SS + (s - 1)]
                                  : &Xn[((size_t)b * TT + tn) * SS + (s - 1 - SS)];
      async_b32(lds_off(xb + (cur ^ 1) * BT + tid), gp);
    }
#pragma unroll
    for (int mt = 0; mt < 4; ++mt) {
      v16h a0 = load_afrag(hb, HLD, mt * 16, 0, lane);
      v16h a1 = load_afrag(hb, HLD, mt * 16, 32, lane);
      v8f  c0 = {0, 0, 0, 0, 0, 0, 0, 0};
      v8f  c1 = {0, 0, 0, 0, 0, 0, 0, 0};
      c0 = WMMA(a0, bf00, c0); c0 = WMMA(a1, bf01, c0);
      c1 = WMMA(a0, bf10, c1); c1 = WMMA(a1, bf11, c1);
      store_cfrag(g, GLD, mt * 16, (wave * 2 + 0) * 16, lane, c0);
      store_cfrag(g, GLD, mt * 16, (wave * 2 + 1) * 16, lane, c1);
    }
    __syncthreads();  // g ready
    const int j = tid & 63;
#pragma unroll
    for (int i = 0; i < 16; ++i) {
      int   b  = i * 4 + (tid >> 6);
      float xv = xb[cur * BT + b];
      float gi = g[b * GLD + j      ] + xv * Wb[j      ] + bb[j      ];
      float gf = g[b * GLD + j +  64] + xv * Wb[j +  64] + bb[j +  64];
      float gg = g[b * GLD + j + 128] + xv * Wb[j + 128] + bb[j + 128];
      float go = g[b * GLD + j + 192] + xv * Wb[j + 192] + bb[j + 192];
      float c  = sigf(gf) * creg[i] + sigf(gi) * tanhfast(gg);
      creg[i]  = c;
      float h  = sigf(go) * tanhfast(c);
      hb[b * HLD + j] = (h16)h;
      size_t o = ((size_t)t * BB + b0 + b) * HH + j;
      if (s == 0)
        outp[o] = h;
      else
        __hip_atomic_fetch_add(&outp[o], h * 0.1f, __ATOMIC_RELAXED,
                               __HIP_MEMORY_SCOPE_AGENT);
    }
  }
}

// ---------------------------------------------------------------- phase B: MI-LSTM scan
struct MiP {
  const float* mats[17];  // Wcy,Wiy,Wf,Wo, Ucy,Uiy,Uf,Uo, Wcp,Wip, Ucp,Uip, Wcn,Win, Ucn,Uin, Wa
  const float* bcy; const float* biy; const float* bf;  const float* bo;
  const float* bcp; const float* bip; const float* bcn; const float* bin;
  const float* va;
};

#define OFF_WALL 0        // 1088*66*2 = 143616
#define OFF_G    143616   // 64*260*4  = 66560  (reused as [192][68] f32 for attention)
#define OFF_FF   210176   // 4096 h16  = 8192
#define OFF_OO   218368   // 8192
#define OFF_HB   226560   // 64*72*2   = 9216
#define OFF_YB0  235776   // 9216  } buffer set 0
#define OFF_PB0  244992   // 9216  }
#define OFF_NB0  254208   // 9216  }
#define OFF_YB1  263424   // 9216  } buffer set 1
#define OFF_PB1  272640   // 9216  }
#define OFF_NB1  281856   // 9216  }
#define BUFSTRIDE 27648   // OFF_YB1 - OFF_YB0
#define OFF_LB   291072   // 192*72*2  = 27648
#define OFF_UV   318720   // 768
#define OFF_BBY  319488   // 1024
#define OFF_BBP  320512   // 512
#define OFF_BBN  321024   // 512
#define OFF_VA   321536   // 256
#define SMEM_B   321792

__global__ void milstm_kernel(MiP p, const h16* __restrict__ yseq16,
                              const h16* __restrict__ Pt16, const h16* __restrict__ Nt16,
                              float* __restrict__ Hs) {
  extern __shared__ char smem[];
  h16*   wAll = (h16*)(smem + OFF_WALL);
  float* g    = (float*)(smem + OFF_G);
  h16*   ff   = (h16*)(smem + OFF_FF);
  h16*   oo   = (h16*)(smem + OFF_OO);
  h16*   hb   = (h16*)(smem + OFF_HB);
  h16*   lb   = (h16*)(smem + OFF_LB);
  float* uv   = (float*)(smem + OFF_UV);
  float* bbY  = (float*)(smem + OFF_BBY);
  float* bbP  = (float*)(smem + OFF_BBP);
  float* bbN  = (float*)(smem + OFF_BBN);
  float* vaS  = (float*)(smem + OFF_VA);

  const int tid = threadIdx.x, lane = tid & 31, wave = tid >> 5;
  const int b0  = blockIdx.x * BT;

  for (int idx = tid; idx < 17 * 4096; idx += 256) {
    int mat = idx >> 12, e = idx & 4095, k = e >> 6, n = e & 63;
    wAll[(mat * 64 + n) * WTLD + k] = (h16)p.mats[mat][e];
  }
  if (tid < 64) {
    bbY[tid] = p.bcy[tid]; bbY[64 + tid] = p.biy[tid];
    bbY[128 + tid] = p.bf[tid]; bbY[192 + tid] = p.bo[tid];
    bbP[tid] = p.bcp[tid]; bbP[64 + tid] = p.bip[tid];
    bbN[tid] = p.bcn[tid]; bbN[64 + tid] = p.bin[tid];
    vaS[tid] = p.va[tid];
  }
  for (int idx = tid; idx < BT * HLD; idx += 256) hb[idx] = (h16)0.f;
  float creg[16];
#pragma unroll
  for (int i = 0; i < 16; ++i) creg[i] = 0.f;
  const int j = tid & 63;

  // prologue: async prefetch tiles for t=0 into buffer set 0
  {
    size_t base = (size_t)b0 * HH;
    issue_tile_async(yseq16 + base, (h16*)(smem + OFF_YB0), tid);
    issue_tile_async(Pt16 + base, (h16*)(smem + OFF_PB0), tid);
    issue_tile_async(Nt16 + base, (h16*)(smem + OFF_NB0), tid);
  }

#pragma unroll 1
  for (int t = 0; t < TT; ++t) {
    wait_async0();
    __syncthreads();  // tiles[t] landed; prev update done
    const int cur = t & 1;
    h16* ycur = (h16*)(smem + OFF_YB0 + cur * BUFSTRIDE);
    h16* pcur = (h16*)(smem + OFF_PB0 + cur * BUFSTRIDE);
    h16* ncur = (h16*)(smem + OFF_NB0 + cur * BUFSTRIDE);
    if (t + 1 < TT) {  // prefetch next step's tiles into the other buffer set
      const int nxt = cur ^ 1;
      size_t base = ((size_t)(t + 1) * BB + b0) * HH;
      issue_tile_async(yseq16 + base, (h16*)(smem + OFF_YB0 + nxt * BUFSTRIDE), tid);
      issue_tile_async(Pt16 + base, (h16*)(smem + OFF_PB0 + nxt * BUFSTRIDE), tid);
      issue_tile_async(Nt16 + base, (h16*)(smem + OFF_NB0 + nxt * BUFSTRIDE), tid);
    }
    // gy = y@[Wcy|Wiy|Wf|Wo] + h@[Ucy|Uiy|Uf|Uo]
    mm_block(wAll, 0, ycur, 256, hb, g, GLD, BT, 256, wave, lane);
    __syncthreads();
#pragma unroll
    for (int i = 0; i < 16; ++i) {
      int   b  = i * 4 + (tid >> 6);
      float Cy = tanhfast(g[b * GLD + j] + bbY[j]);
      float iy = sigf(g[b * GLD + 64 + j] + bbY[64 + j]);
      ff[b * 64 + j] = (h16)sigf(g[b * GLD + 128 + j] + bbY[128 + j]);
      oo[b * 64 + j] = (h16)sigf(g[b * GLD + 192 + j] + bbY[192 + j]);
      lb[b * HLD + j] = (h16)(Cy * iy);
    }
    __syncthreads();
    mm_block(wAll, 512, pcur, 640, hb, g, GLD, BT, 128, wave, lane);
    __syncthreads();
#pragma unroll
    for (int i = 0; i < 16; ++i) {
      int   b  = i * 4 + (tid >> 6);
      float Cp = tanhfast(g[b * GLD + j] + bbP[j]);
      float ip = sigf(g[b * GLD + 64 + j] + bbP[64 + j]);
      lb[(64 + b) * HLD + j] = (h16)(Cp * ip);
    }
    __syncthreads();
    mm_block(wAll, 768, ncur, 896, hb, g, GLD, BT, 128, wave, lane);
    __syncthreads();
#pragma unroll
    for (int i = 0; i < 16; ++i) {
      int   b   = i * 4 + (tid >> 6);
      float Cn  = tanhfast(g[b * GLD + j] + bbN[j]);
      float in_ = sigf(g[b * GLD + 64 + j] + bbN[64 + j]);
      lb[(128 + b) * HLD + j] = (h16)(Cn * in_);
    }
    __syncthreads();
    // attention scores: [192,64] = l @ Wa  (Wa rows at 1024)
    mm_block(wAll, 1024, lb, 0, (const h16*)nullptr, g, 68, 192, 64, wave, lane);
    __syncthreads();
    if (tid < 192) {
      float acc = 0.f;
      const float* row = g + tid * 68;
      for (int h = 0; h < HH; ++h) acc += tanhfast(row[h]) * vaS[h];
      uv[tid] = acc;
    }
    __syncthreads();
#pragma unroll
    for (int i = 0; i < 16; ++i) {
      int   b  = i * 4 + (tid >> 6);
      float u0 = uv[b], u1 = uv[64 + b], u2 = uv[128 + b];
      float mx = fmaxf(u0, fmaxf(u1, u2));
      float e0 = __expf(u0 - mx), e1 = __expf(u1 - mx), e2 = __expf(u2 - mx);
      float inv = 1.f / (e0 + e1 + e2);
      float L = (e0 * (float)lb[b * HLD + j] + e1 * (float)lb[(64 + b) * HLD + j] +
                 e2 * (float)lb[(128 + b) * HLD + j]) * inv;
      float c = (float)ff[b * 64 + j] * creg[i] + L;
      creg[i] = c;
      float h = (float)oo[b * 64 + j] * tanhfast(c);
      hb[b * HLD + j] = (h16)h;
      Hs[((size_t)t * BB + b0 + b) * HH + j] = h;
    }
  }
}

// ---------------------------------------------------------------- phase C: temporal attention
__global__ void attn_e_kernel(const float* __restrict__ Hs, const float* __restrict__ Wq,
                              const float* __restrict__ bq, const float* __restrict__ vq,
                              float* __restrict__ e) {
  int gid = blockIdx.x * 256 + threadIdx.x;  // B*T threads
  int b = gid >> 7, t = gid & 127;
  float hrow[HH];
  const float* src = Hs + ((size_t)t * BB + b) * HH;
#pragma unroll
  for (int k = 0; k < HH; ++k) hrow[k] = src[k];
  float acc = 0.f;
  for (int jj = 0; jj < HH; ++jj) {
    float s = bq[jj];
    for (int k = 0; k < HH; ++k) s += hrow[k] * Wq[k * HH + jj];
    acc += tanhfast(s) * vq[jj];
  }
  e[(size_t)b * TT + t] = acc;
}

__global__ void attn_out_kernel(const float* __restrict__ Hs, const float* __restrict__ e,
                                const float* __restrict__ Wl, const float* __restrict__ bl,
                                float* __restrict__ out) {
  __shared__ float beta_s[8][TT];
  const int lane = threadIdx.x & 31, wave = threadIdx.x >> 5;
  const int b = blockIdx.x * 8 + wave;
  float v[4];
  float mx = -3.4e38f;
#pragma unroll
  for (int q = 0; q < 4; ++q) {
    v[q] = e[(size_t)b * TT + lane + q * 32];
    mx = fmaxf(mx, v[q]);
  }
  for (int o = 16; o; o >>= 1) mx = fmaxf(mx, __shfl_xor(mx, o, 32));
  float sum = 0.f;
#pragma unroll
  for (int q = 0; q < 4; ++q) { v[q] = __expf(v[q] - mx); sum += v[q]; }
  for (int o = 16; o; o >>= 1) sum += __shfl_xor(sum, o, 32);
  float inv = 1.f / sum;
#pragma unroll
  for (int q = 0; q < 4; ++q) beta_s[wave][lane + q * 32] = v[q] * inv;
  __syncthreads();
  const int j0 = lane * 2, j1 = j0 + 1;
  float c0 = 0.f, c1 = 0.f;
  for (int t = 0; t < TT; ++t) {
    float bt = beta_s[wave][t];
    const float* hp = Hs + ((size_t)t * BB + b) * HH;
    c0 += bt * hp[j0];
    c1 += bt * hp[j1];
  }
  float part = c0 * Wl[j0] + c1 * Wl[j1];
  for (int o = 16; o; o >>= 1) part += __shfl_xor(part, o, 32);
  if (lane == 0) out[b] = fmaxf(part + bl[0], 0.f);
}

// ---------------------------------------------------------------- launch
extern "C" void kernel_launch(void* const* d_in, const int* in_sizes, int n_in,
                              void* d_out, int out_size, void* d_ws, size_t ws_size,
                              hipStream_t stream) {
  (void)in_sizes; (void)n_in; (void)out_size; (void)ws_size;
  // setup_inputs order: Y, X_n, X_p, then params in make_params insertion order
  const float* Y   = (const float*)d_in[0];
  const float* Xn  = (const float*)d_in[1];
  const float* Xp  = (const float*)d_in[2];
  const float* Wy  = (const float*)d_in[3];
  const float* Uy  = (const float*)d_in[4];
  const float* by  = (const float*)d_in[5];
  const float* Wp  = (const float*)d_in[6];
  const float* Up  = (const float*)d_in[7];
  const float* bp  = (const float*)d_in[8];
  const float* Wn  = (const float*)d_in[9];
  const float* Un  = (const float*)d_in[10];
  const float* bn  = (const float*)d_in[11];
  const float* Wcy = (const float*)d_in[12];
  const float* Ucy = (const float*)d_in[13];
  const float* bcy = (const float*)d_in[14];
  const float* Wcp = (const float*)d_in[15];
  const float* Ucp = (const float*)d_in[16];
  const float* bcp = (const float*)d_in[17];
  const float* Wcn = (const float*)d_in[18];
  const float* Ucn = (const float*)d_in[19];
  const float* bcn = (const float*)d_in[20];
  const float* Wiy = (const float*)d_in[21];
  const float* Uiy = (const float*)d_in[22];
  const float* biy = (const float*)d_in[23];
  const float* Wip = (const float*)d_in[24];
  const float* Uip = (const float*)d_in[25];
  const float* bip = (const float*)d_in[26];
  const float* Win = (const float*)d_in[27];
  const float* Uin = (const float*)d_in[28];
  const float* bin = (const float*)d_in[29];
  const float* Wf  = (const float*)d_in[30];
  const float* Uf  = (const float*)d_in[31];
  const float* bf  = (const float*)d_in[32];
  const float* Wo  = (const float*)d_in[33];
  const float* Uo  = (const float*)d_in[34];
  const float* bo  = (const float*)d_in[35];
  const float* Wa  = (const float*)d_in[36];
  const float* va  = (const float*)d_in[37];
  const float* Wq  = (const float*)d_in[38];
  const float* bq  = (const float*)d_in[39];
  const float* vq  = (const float*)d_in[40];
  const float* Wl  = (const float*)d_in[41];
  const float* bl  = (const float*)d_in[42];

  const size_t TBH = (size_t)TT * BB * HH;  // 8388608
  float* ws    = (float*)d_ws;
  float* yseq  = ws;                 // f32 [T][B][H]
  float* Pt    = yseq + TBH;
  float* Nt    = Pt + TBH;
  float* Hs    = Nt + TBH;
  float* ebuf  = Hs + TBH;
  h16*   f16b  = (h16*)(ebuf + (size_t)BB * TT);  // f16 copies: yseq16|Pt16|Nt16
  h16*   yseq16 = f16b;
  h16*   Pt16   = f16b + TBH;
  h16*   Nt16   = f16b + 2 * TBH;

  zero_kernel<<<4096, 256, 0, stream>>>(Pt, (int)(2 * TBH));

  lstm_scan_kernel<<<dim3(NTILES, 1 + 2 * SS), 256, SMEM_A, stream>>>(
      Y, Xp, Xn, Wy, Uy, by, Wp, Up, bp, Wn, Un, bn, yseq, Pt, Nt);

  // convert yseq|Pt|Nt (contiguous) to f16 for WMMA + async-LDS consumption
  cvt_f16_kernel<<<4096, 256, 0, stream>>>(yseq, yseq16, (int)(3 * TBH / 2));

  MiP mp;
  const float* mats[17] = {Wcy, Wiy, Wf, Wo, Ucy, Uiy, Uf, Uo, Wcp, Wip,
                           Ucp, Uip, Wcn, Win, Ucn, Uin, Wa};
  for (int i = 0; i < 17; ++i) mp.mats[i] = mats[i];
  mp.bcy = bcy; mp.biy = biy; mp.bf = bf; mp.bo = bo;
  mp.bcp = bcp; mp.bip = bip; mp.bcn = bcn; mp.bin = bin; mp.va = va;

  milstm_kernel<<<NTILES, 256, SMEM_B, stream>>>(mp, yseq16, Pt16, Nt16, Hs);

  attn_e_kernel<<<(BB * TT) / 256, 256, 0, stream>>>(Hs, Wq, bq, vq, ebuf);
  attn_out_kernel<<<BB / 8, 256, 0, stream>>>(Hs, ebuf, Wl, bl, (float*)d_out);
}
